// MultiheadAttention_71382356459799
// MI455X (gfx1250) — matrix operand order repaired and checked by reading, not compiled, not run on hardware
//
#include <hip/hip_runtime.h>
#include <hip/hip_bf16.h>

// Problem constants
#define TT 2048
#define BB 4
#define EE 512
#define HH 8
#define HD 64
#define SCALE 0.125f  // 1/sqrt(64)

typedef __attribute__((ext_vector_type(16))) __bf16 v16bf;
typedef __attribute__((ext_vector_type(8)))  __bf16 v8bf;
typedef __attribute__((ext_vector_type(8)))  float  v8f;
typedef __attribute__((ext_vector_type(4)))  unsigned v4u;
typedef __attribute__((ext_vector_type(8)))  unsigned v8u;

__device__ inline v8f wmma_bf16(v16bf a, v16bf b, v8f c) {
  // D = A(16x32 bf16) * B(32x16 bf16) + C(16x16 f32)
  return __builtin_amdgcn_wmma_f32_16x16x32_bf16(
      /*neg_a=*/false, a, /*neg_b=*/false, b,
      /*c_mod=*/(short)0, c, /*reuse_a=*/false, /*reuse_b=*/false);
}

// ---- Fragment loaders (CDNA5 wave32 layouts, CDNA5 ISA §7.12.2) ----

// A: 16x32 (MxK) bf16 tile from row-major src (leading dim ld, elements).
__device__ inline v16bf load_a_frag(const __bf16* src, int ld, int lane) {
  int m = lane & 15, g = lane >> 4;
  v16bf a;
#pragma unroll
  for (int v = 0; v < 8; ++v) {
    int kb = ((v < 4) ? (2 * v) : (16 + 2 * (v - 4))) + 8 * g;
    a[2 * v]     = src[m * ld + kb];
    a[2 * v + 1] = src[m * ld + kb + 1];
  }
  return a;
}

// Same, but source is fp32 (probabilities in LDS) converted to bf16 on the fly.
__device__ inline v16bf load_a_frag_f32(const float* src, int ld, int lane) {
  int m = lane & 15, g = lane >> 4;
  v16bf a;
#pragma unroll
  for (int v = 0; v < 8; ++v) {
    int kb = ((v < 4) ? (2 * v) : (16 + 2 * (v - 4))) + 8 * g;
    a[2 * v]     = (__bf16)src[m * ld + kb];
    a[2 * v + 1] = (__bf16)src[m * ld + kb + 1];
  }
  return a;
}

// B: 32x16 (KxN) tile where source is N-major (row-major [N][K]).
// Lane l: N = l&15; lanes 0-15 hold K=0..7,16..23, lanes 16-31 hold K=8..15,24..31.
// Two runs of 8 contiguous bf16 (16 bytes) per lane -> 2x b128 loads.
__device__ inline v16bf load_bT_frag(const __bf16* src, int ld, int lane) {
  int n = lane & 15, kb = (lane >> 4) * 8;
  v16bf b;
#pragma unroll
  for (int i = 0; i < 16; ++i) b[i] = src[n * ld + kb + ((i < 8) ? i : (16 + i - 8))];
  return b;
}

// ---- Tensor Data Mover helpers (CDNA5 ISA-8) --------

// LDS byte offset of a __shared__ object (generic -> addrspace(3) -> offset).
__device__ inline unsigned lds_off(void* p) {
  return (unsigned)(uintptr_t)(__attribute__((address_space(3))) void*)p;
}

// D# group1 for a 64x32 bf16 tile out of a [tensor_rows][512] bf16 tensor,
// with LDS padding of 4 DWORDs (8 bf16) after every 16 DWORDs (32 bf16 row)
// so the tile lands in a [64][40] padded LDS array.
__device__ inline v8u tdm_g1_64x32(unsigned tensor_rows) {
  v8u g;
  // word0: wg_mask=0 | data_size=1(2B)<<16 | pad_enable(bit20)
  //        | pad_interval=3 (16 DW) <<22 | pad_amount=3 (4 DW) <<25
  g[0] = 0x00010000u | 0x00100000u | (3u << 22) | (3u << 25);
  g[1] = (unsigned)EE << 16;                    // tensor_dim0[15:0] = 512
  g[2] = (tensor_rows & 0xFFFFu) << 16;         // tensor_dim1[15:0]
  g[3] = ((tensor_rows >> 16) & 0xFFFFu) | (32u << 16);  // dim1 hi | tile_dim0=32
  g[4] = 64u;                                   // tile_dim1 = 64 (tile_dim2 = 0)
  g[5] = (unsigned)EE;                          // tensor_dim0_stride = 512
  g[6] = 0u;                                    // dim0_stride hi | dim1_stride lo
  g[7] = 0u;                                    // dim1_stride hi
  return g;
}

// Issue TENSOR_LOAD_TO_LDS with a 2-group descriptor (2D tile).
__device__ inline void tdm_load(unsigned lds_addr, const void* gaddr, v8u g1) {
  unsigned long long ga = (unsigned long long)(uintptr_t)gaddr;
  v4u g0;
  g0[0] = 1u;                                   // count=1, user descriptor
  g0[1] = lds_addr;                             // lds_addr (bytes)
  g0[2] = (unsigned)ga;                         // global_addr[31:0]
  g0[3] = ((unsigned)(ga >> 32) & 0x01FFFFFFu) | (2u << 30);  // addr[56:32]|type=2
  asm volatile("tensor_load_to_lds %0, %1" :: "s"(g0), "s"(g1) : "memory");
}

// ---------------------------------------------------------------------------
// Kernel 1: fp32 -> bf16 conversion, 8 elements per thread (b128 traffic).
// ---------------------------------------------------------------------------
__global__ __launch_bounds__(256) void cvt_f32_bf16(const float* __restrict__ in,
                                                    __bf16* __restrict__ out, int n) {
  int i = (blockIdx.x * 256 + threadIdx.x) * 8;
  if (i + 8 <= n) {
    v8f f = *(const v8f*)(in + i);
    v8bf o;
#pragma unroll
    for (int j = 0; j < 8; ++j) o[j] = (__bf16)f[j];
    *(v8bf*)(out + i) = o;
  }
}

// ---------------------------------------------------------------------------
// Kernel 2: QKV projection. qkv[m,n] = sum_k x[m,k]*W[n,k] + bias[n]
// M = T*B = 8192, N = 3E = 1536, K = E = 512.
// A/B panels DMA'd into LDS by the Tensor Data Mover (TENSORcnt).
// q,k stored [B*H][T][HD]; v stored TRANSPOSED [B*H][HD][T] for the PV GEMM.
// ---------------------------------------------------------------------------
__global__ __launch_bounds__(128) void qkv_gemm(const __bf16* __restrict__ xb,
                                                const __bf16* __restrict__ wb,
                                                const float* __restrict__ bias,
                                                __bf16* __restrict__ qb,
                                                __bf16* __restrict__ kb,
                                                __bf16* __restrict__ vtb) {
  __shared__ __bf16 As[64][40];  // 64 M-rows x 32 K (+4 DW pad, matches TDM pad)
  __shared__ __bf16 Bs[64][40];  // 64 N-rows x 32 K (+pad)

  int m0 = blockIdx.y * 64;
  int n0 = blockIdx.x * 64;
  int tid = threadIdx.x, lane = tid & 31, wave = tid >> 5;
  int msub = (wave >> 1) * 32, nsub = (wave & 1) * 32;

  v8u g1x = tdm_g1_64x32(TT * BB);      // x is [8192][512]
  v8u g1w = tdm_g1_64x32(3 * EE);       // W is [1536][512]
  unsigned ldsA = lds_off(&As[0][0]);
  unsigned ldsB = lds_off(&Bs[0][0]);

  v8f acc[2][2] = {};

  for (int k0 = 0; k0 < EE; k0 += 32) {
    if (tid < 32) {  // one wave drives the DMA engine
      tdm_load(ldsA, xb + (size_t)m0 * EE + k0, g1x);
      tdm_load(ldsB, wb + (size_t)n0 * EE + k0, g1w);
      __builtin_amdgcn_s_wait_tensorcnt(0);
    }
    __syncthreads();
    v16bf a0 = load_a_frag(&As[msub][0], 40, lane);
    v16bf a1 = load_a_frag(&As[msub + 16][0], 40, lane);
    v16bf b0 = load_bT_frag(&Bs[nsub][0], 40, lane);
    v16bf b1 = load_bT_frag(&Bs[nsub + 16][0], 40, lane);
    acc[0][0] = wmma_bf16(a0, b0, acc[0][0]);
    acc[0][1] = wmma_bf16(a0, b1, acc[0][1]);
    acc[1][0] = wmma_bf16(a1, b0, acc[1][0]);
    acc[1][1] = wmma_bf16(a1, b1, acc[1][1]);
    __syncthreads();
  }

  int nl = lane & 15, g = lane >> 4;
#pragma unroll
  for (int ti = 0; ti < 2; ++ti)
#pragma unroll
    for (int tj = 0; tj < 2; ++tj)
#pragma unroll
      for (int r = 0; r < 8; ++r) {
        int m = m0 + msub + ti * 16 + r + 8 * g;   // row of x: m = t*B + b
        int nn = n0 + nsub + tj * 16 + nl;         // col in [0, 3E)
        float v = acc[ti][tj][r] + bias[nn];
        int which = nn >> 9;                       // 0=q, 1=k, 2=v
        int e = nn & 511;
        int h = e >> 6, d = e & 63;
        int t = m >> 2, b = m & 3;                 // B == 4
        size_t bh = (size_t)(b * HH + h);
        if (which == 0)
          qb[(bh * TT + t) * HD + d] = (__bf16)v;
        else if (which == 1)
          kb[(bh * TT + t) * HD + d] = (__bf16)v;
        else
          vtb[(bh * HD + d) * TT + t] = (__bf16)v;  // V^T: [HD][T]
      }
}

// ---------------------------------------------------------------------------
// Kernel 3: attention. One block per (b, 16-row q tile); loops over 8 heads.
// Dynamic LDS: S[16][SLD] (scores/probs, fp32) + AVG[16][SLD] (head-avg probs).
// ---------------------------------------------------------------------------
#define SLD 2056

__global__ __launch_bounds__(128) void attn_kernel(const __bf16* __restrict__ qb,
                                                   const __bf16* __restrict__ kb,
                                                   const __bf16* __restrict__ vtb,
                                                   __bf16* __restrict__ ctxb,
                                                   float* __restrict__ avg_out) {
  extern __shared__ float smem[];
  float* S   = smem;                 // [16][SLD]
  float* AVG = smem + 16 * SLD;      // [16][SLD]
  __shared__ float red[16][8];
  __shared__ float mrow[16];
  __shared__ float lrow[16];

  int qt = blockIdx.x;        // 0..127
  int b  = blockIdx.y;        // 0..3
  int t0 = qt * 16;
  int tid = threadIdx.x, lane = tid & 31, wave = tid >> 5;
  int nl = lane & 15, g = lane >> 4;
  int row = tid >> 3, sub = tid & 7;   // softmax: 8 threads per q-row

  for (int h = 0; h < HH; ++h) {
    const __bf16* Q  = qb  + ((size_t)(b * HH + h) * TT + t0) * HD;
    const __bf16* K  = kb  + (size_t)(b * HH + h) * TT * HD;   // [T][HD]
    const __bf16* Vt = vtb + (size_t)(b * HH + h) * HD * TT;   // [HD][T]

    // ---- scores: S = (Q K^T) * SCALE -----------------------------------
    v16bf aq0 = load_a_frag(Q, HD, lane);        // k-dim 0..31
    v16bf aq1 = load_a_frag(Q + 32, HD, lane);   // k-dim 32..63
    for (int kt = wave; kt < TT / 16; kt += 4) {
      int kcol = kt * 16;
      if (kcol + 64 < TT)
        __builtin_prefetch(K + (size_t)(kcol + 64) * HD, 0, 1);
      // B = K^T (HD x keys): element (kd, key) = K[key][kd]  -> N-major source
      v16bf bk0 = load_bT_frag(K + (size_t)kcol * HD, HD, lane);
      v16bf bk1 = load_bT_frag(K + (size_t)kcol * HD + 32, HD, lane);
      v8f acc = {};
      acc = wmma_bf16(aq0, bk0, acc);
      acc = wmma_bf16(aq1, bk1, acc);
#pragma unroll
      for (int r = 0; r < 8; ++r)
        S[(r + 8 * g) * SLD + kcol + nl] = acc[r] * SCALE;
    }
    __syncthreads();

    // ---- softmax (fp32, two sweeps) ------------------------------------
    float mloc = -3.0e38f;
    for (int c = sub; c < TT; c += 8) mloc = fmaxf(mloc, S[row * SLD + c]);
    red[row][sub] = mloc;
    __syncthreads();
    if (sub == 0) {
      float mm = red[row][0];
#pragma unroll
      for (int i = 1; i < 8; ++i) mm = fmaxf(mm, red[row][i]);
      mrow[row] = mm;
    }
    __syncthreads();
    float m = mrow[row], sloc = 0.0f;
    for (int c = sub; c < TT; c += 8) {
      float p = __expf(S[row * SLD + c] - m);
      S[row * SLD + c] = p;         // unnormalized prob
      sloc += p;
    }
    red[row][sub] = sloc;
    __syncthreads();
    if (sub == 0) {
      float ss = 0.0f;
#pragma unroll
      for (int i = 0; i < 8; ++i) ss += red[row][i];
      lrow[row] = ss;
    }
    __syncthreads();

    // accumulate normalized probs / H into AVG strip
    float invl = 1.0f / (lrow[row] * (float)HH);
    if (h == 0) {
      for (int c = sub; c < TT; c += 8) AVG[row * SLD + c] = S[row * SLD + c] * invl;
    } else {
      for (int c = sub; c < TT; c += 8) AVG[row * SLD + c] += S[row * SLD + c] * invl;
    }
    __syncthreads();   // all probs final before P@V reads whole strip

    // ---- O = P @ V : each wave owns one 16-wide slice of HD ------------
    int d0 = wave * 16;
    v8f oacc = {};
    for (int kk = 0; kk < TT; kk += 32) {
      if (kk + 128 < TT)
        __builtin_prefetch(Vt + (size_t)d0 * TT + kk + 128, 0, 1);
      v16bf ap = load_a_frag_f32(S + kk, SLD, lane);
      // B[k=key][n=d] = V[key][d] = Vt[d][key]  -> N-major source, ld = T
      v16bf bv = load_bT_frag(Vt + (size_t)d0 * TT + kk, TT, lane);
      oacc = wmma_bf16(ap, bv, oacc);
    }
#pragma unroll
    for (int r = 0; r < 8; ++r) {
      int M = r + 8 * g;
      float o = oacc[r] / lrow[M];
      // ctx layout [T][B][E], e = h*HD + d
      ctxb[((size_t)(t0 + M) * BB + b) * EE + h * HD + d0 + nl] = (__bf16)o;
    }
    __syncthreads();   // S/AVG strips reused by next head
  }

  // ---- write avg_weights strip: avg_out[b][t0+r][k] ----------------------
  for (int idx = tid; idx < 16 * TT; idx += 128) {
    int r = idx >> 11;          // /2048
    int c = idx & (TT - 1);
    avg_out[((size_t)b * TT + (t0 + r)) * TT + c] = AVG[r * SLD + c];
  }
}

// ---------------------------------------------------------------------------
// Kernel 4: out projection. out[m,n] = sum_k ctx[m,k]*OW[n,k] + out_b[n], fp32.
// Panels DMA'd by TDM, same as qkv_gemm.
// ---------------------------------------------------------------------------
__global__ __launch_bounds__(128) void out_gemm(const __bf16* __restrict__ ctxb,
                                                const __bf16* __restrict__ owb,
                                                const float* __restrict__ ob,
                                                float* __restrict__ out) {
  __shared__ __bf16 As[64][40];
  __shared__ __bf16 Bs[64][40];

  int m0 = blockIdx.y * 64;
  int n0 = blockIdx.x * 64;
  int tid = threadIdx.x, lane = tid & 31, wave = tid >> 5;
  int msub = (wave >> 1) * 32, nsub = (wave & 1) * 32;

  v8u g1c = tdm_g1_64x32(TT * BB);      // ctx is [8192][512]
  v8u g1w = tdm_g1_64x32(EE);           // out_w is [512][512]
  unsigned ldsA = lds_off(&As[0][0]);
  unsigned ldsB = lds_off(&Bs[0][0]);

  v8f acc[2][2] = {};

  for (int k0 = 0; k0 < EE; k0 += 32) {
    if (tid < 32) {
      tdm_load(ldsA, ctxb + (size_t)m0 * EE + k0, g1c);
      tdm_load(ldsB, owb + (size_t)n0 * EE + k0, g1w);
      __builtin_amdgcn_s_wait_tensorcnt(0);
    }
    __syncthreads();
    v16bf a0 = load_a_frag(&As[msub][0], 40, lane);
    v16bf a1 = load_a_frag(&As[msub + 16][0], 40, lane);
    v16bf b0 = load_bT_frag(&Bs[nsub][0], 40, lane);
    v16bf b1 = load_bT_frag(&Bs[nsub + 16][0], 40, lane);
    acc[0][0] = wmma_bf16(a0, b0, acc[0][0]);
    acc[0][1] = wmma_bf16(a0, b1, acc[0][1]);
    acc[1][0] = wmma_bf16(a1, b0, acc[1][0]);
    acc[1][1] = wmma_bf16(a1, b1, acc[1][1]);
    __syncthreads();
  }

  int nl = lane & 15, g = lane >> 4;
#pragma unroll
  for (int ti = 0; ti < 2; ++ti)
#pragma unroll
    for (int tj = 0; tj < 2; ++tj)
#pragma unroll
      for (int r = 0; r < 8; ++r) {
        int m = m0 + msub + ti * 16 + r + 8 * g;
        int nn = n0 + nsub + tj * 16 + nl;
        out[(size_t)m * EE + nn] = acc[ti][tj][r] + ob[nn];
      }
}

// ---------------------------------------------------------------------------
extern "C" void kernel_launch(void* const* d_in, const int* in_sizes, int n_in,
                              void* d_out, int out_size, void* d_ws, size_t ws_size,
                              hipStream_t stream) {
  (void)in_sizes; (void)n_in; (void)out_size; (void)ws_size;

  const float* x    = (const float*)d_in[0];  // [T,B,E]
  const float* w_in = (const float*)d_in[1];  // [3E,E]
  const float* b_in = (const float*)d_in[2];  // [3E]
  const float* w_o  = (const float*)d_in[3];  // [E,E]
  const float* b_o  = (const float*)d_in[4];  // [E]
  float* out = (float*)d_out;                 // attn [T,B,E] ++ avg [B,T,T]

  // workspace layout (bf16 buffers)
  const size_t N_X    = (size_t)TT * BB * EE;        // 4,194,304
  const size_t N_WQKV = (size_t)3 * EE * EE;         // 786,432
  const size_t N_OW   = (size_t)EE * EE;             // 262,144
  const size_t N_HEAD = (size_t)BB * HH * TT * HD;   // 4,194,304 per q/k/v

  char* ws = (char*)d_ws;
  __bf16* xb   = (__bf16*)ws;            ws += N_X * 2;
  __bf16* wqkv = (__bf16*)ws;            ws += N_WQKV * 2;
  __bf16* owb  = (__bf16*)ws;            ws += N_OW * 2;
  __bf16* qb   = (__bf16*)ws;            ws += N_HEAD * 2;
  __bf16* kb   = (__bf16*)ws;            ws += N_HEAD * 2;
  __bf16* vtb  = (__bf16*)ws;            ws += N_HEAD * 2;  // V^T per head
  __bf16* ctxb = (__bf16*)ws;            ws += N_X * 2;

  // 1) convert inputs to bf16 (8 elements per thread)
  cvt_f32_bf16<<<dim3(N_X / 8 / 256), 256, 0, stream>>>(x, xb, (int)N_X);
  cvt_f32_bf16<<<dim3(N_WQKV / 8 / 256), 256, 0, stream>>>(w_in, wqkv, (int)N_WQKV);
  cvt_f32_bf16<<<dim3(N_OW / 8 / 256), 256, 0, stream>>>(w_o, owb, (int)N_OW);

  // 2) QKV projection: M=8192, N=1536 -> grid (24, 128)
  qkv_gemm<<<dim3(3 * EE / 64, TT * BB / 64), 128, 0, stream>>>(xb, wqkv, b_in, qb, kb, vtb);

  // 3) attention: (q-tile, batch), 257 KB dynamic LDS per block
  size_t smem_bytes = (size_t)2 * 16 * SLD * sizeof(float);
  float* avg_out = out + N_X;
  attn_kernel<<<dim3(TT / 16, BB), 128, smem_bytes, stream>>>(qb, kb, vtb, ctxb, avg_out);

  // 4) out projection: M=8192, N=512 -> grid (8, 128)
  out_gemm<<<dim3(EE / 64, TT * BB / 64), 128, 0, stream>>>(ctxb, owb, b_o, out);
}
